// TorchMD_S_20907900797526
// MI455X (gfx1250) — compile-verified, matrix operand order checked by reading, not run on hardware
//
#include <hip/hip_runtime.h>
#include <hip/hip_bf16.h>
#include <math.h>

#define N_NODES 10000
#define N_EDGES 320000
#define HDIM    128
#define NHEADS  8
#define HEADD   16
#define NRBF    50
#define NRBFP   64
#define NLAYERS 6
#define CUTOFF  5.0f
#define PI_OVER_CUT 0.6283185307179586f

typedef __attribute__((ext_vector_type(16))) __bf16 v16bf;
typedef __attribute__((ext_vector_type(8)))  float  v8f;

// ---------------------------------------------------------------------------
// WMMA fragment helpers (CDNA5 wave32 layouts, 16x16x32 bf16 -> f32)
// A 16x32: lane<16 -> row=lane, K {0..7,16..23}; lane>=16 -> row=lane-16, K {8..15,24..31}
// B is pre-packed fragment-major: ((kt*8 + nt)*32 + lane)*16 contiguous bf16
// C 16x16: VGPR r -> row r (lanes 0-15) / row r+8 (lanes 16-31), col = lane&15
// ---------------------------------------------------------------------------
__device__ __forceinline__ v16bf load_a_frag(const __bf16* p /* &A[row][k0 + hf*8] */) {
    v16bf a;
#pragma unroll
    for (int i = 0; i < 8; ++i) a[i] = p[i];
#pragma unroll
    for (int i = 0; i < 8; ++i) a[8 + i] = p[16 + i];
    return a;
}

__device__ __forceinline__ v16bf load_b_packed(const __bf16* Bp, int kt, int nt, int lane) {
    const __bf16* p = Bp + ((((size_t)kt * 8 + nt) * 32 + lane) << 4);
    v16bf b;
#pragma unroll
    for (int i = 0; i < 16; ++i) b[i] = p[i];
    return b;
}

__device__ __forceinline__ v8f wmma_bf16(v16bf a, v16bf b, v8f c) {
    return __builtin_amdgcn_wmma_f32_16x16x32_bf16(false, a, false, b, (short)0, c, false, false);
}

__device__ __forceinline__ float silu(float x) { return x / (1.f + __expf(-x)); }

// ---------------------------------------------------------------------------
// Utility kernels
// ---------------------------------------------------------------------------
__global__ void k_zero(float* p, int n) {
    int i = blockIdx.x * 256 + threadIdx.x;
    if (i < n) p[i] = 0.f;
}

// Pack fp32 weight (srcRows x 128, zero-padded to Kpad) into WMMA-fragment-major bf16.
// dst element idx: i = idx&15, lane = (idx>>4)&31, nt = (idx>>9)&7, kt = idx>>12
// value = B[kt*32 + hf*8 + i + (i>=8 ? 8 : 0)][nt*16 + (lane&15)], hf = lane>>4
__global__ void k_pack_b(const float* __restrict__ src, __bf16* __restrict__ dst,
                         int srcRows, int Kpad) {
    int idx = blockIdx.x * 256 + threadIdx.x;
    int total = Kpad * HDIM;
    if (idx >= total) return;
    int i    = idx & 15;
    int lane = (idx >> 4) & 31;
    int nt   = (idx >> 9) & 7;
    int kt   = idx >> 12;
    int hf   = lane >> 4;
    int k = kt * 32 + hf * 8 + i + ((i >= 8) ? 8 : 0);
    int n = nt * 16 + (lane & 15);
    dst[idx] = (k < srcRows) ? (__bf16)src[k * HDIM + n] : (__bf16)0.f;
}

// fp32 (rows x cols) -> bf16 into strided destination
__global__ void k_f32_to_bf16_strided(const float* __restrict__ src, __bf16* __restrict__ dst,
                                      int rows, int cols, int dstStride, int dstOff) {
    int i = blockIdx.x * 256 + threadIdx.x;
    if (i >= rows * cols) return;
    int r = i / cols, c = i - r * cols;
    dst[(size_t)r * dstStride + dstOff + c] = (__bf16)src[i];
}

// xc[:, 0:128] = bf16(emb[z])
__global__ void k_embed(const int* __restrict__ z, const float* __restrict__ emb,
                        __bf16* __restrict__ xc) {
    int i = blockIdx.x * 256 + threadIdx.x;
    if (i >= N_NODES * HDIM) return;
    int r = i >> 7, c = i & 127;
    xc[(size_t)r * 256 + c] = (__bf16)emb[z[r] * HDIM + c];
}

// ---------------------------------------------------------------------------
// Edge geometry: distance, cosine cutoff, masked cutoff, expnorm RBF (bf16, padded)
// ---------------------------------------------------------------------------
__global__ void k_edge_geom(const int* __restrict__ ei, const float* __restrict__ pos,
                            const float* __restrict__ means, const float* __restrict__ betas,
                            __bf16* __restrict__ ea, float* __restrict__ C,
                            float* __restrict__ cm) {
    int e = blockIdx.x * 256 + threadIdx.x;
    if (e >= N_EDGES) return;
    int s = ei[e], d_ = ei[N_EDGES + e];
    float dx = pos[d_ * 3 + 0] - pos[s * 3 + 0];
    float dy = pos[d_ * 3 + 1] - pos[s * 3 + 1];
    float dz = pos[d_ * 3 + 2] - pos[s * 3 + 2];
    float d = sqrtf(dx * dx + dy * dy + dz * dz + 1e-12f);
    float cut = (d < CUTOFF) ? 0.5f * (__cosf(d * PI_OVER_CUT) + 1.f) : 0.f;
    C[e]  = cut;
    cm[e] = (s != d_) ? cut : 0.f;
    float ed = __expf(-d);
    __bf16* row = ea + (size_t)e * NRBFP;
#pragma unroll 8
    for (int j = 0; j < NRBFP; ++j) {
        float v = 0.f;
        if (j < NRBF) {
            float t = ed - means[j];
            v = cut * __expf(-betas[j] * t * t);
        }
        row[j] = (__bf16)v;
    }
}

// ---------------------------------------------------------------------------
// Generic GEMM: D[M x 128] = act(A[M x K]_bf16 @ Bp_packed + bias) [+ resid]
// 256 threads = 8 waves; block computes 64 rows; wave w owns column tile w*16.
// ---------------------------------------------------------------------------
__global__ __launch_bounds__(256) void k_gemm128(
    const __bf16* __restrict__ A, int M, int K,
    const __bf16* __restrict__ Bp, const float* __restrict__ bias,
    const float* __restrict__ resid, float* __restrict__ outF,
    __bf16* __restrict__ outB, int act) {
    int lane = threadIdx.x & 31, wave = threadIdx.x >> 5;
    int hf = lane >> 4, lm = lane & 15;
    int n0 = wave * 16;
    int m_base = blockIdx.x * 64;

    v8f acc[4];
#pragma unroll
    for (int rt = 0; rt < 4; ++rt)
#pragma unroll
        for (int r = 0; r < 8; ++r) acc[rt][r] = 0.f;

    for (int k0 = 0; k0 < K; k0 += 32) {
        v16bf bf = load_b_packed(Bp, k0 >> 5, wave, lane);
#pragma unroll
        for (int rt = 0; rt < 4; ++rt) {
            int m = m_base + rt * 16 + lm;
            if (m >= M) m = M - 1;  // clamp; out-of-range results discarded at store
            v16bf af = load_a_frag(A + (size_t)m * K + k0 + hf * 8);
            acc[rt] = wmma_bf16(af, bf, acc[rt]);
        }
    }
#pragma unroll
    for (int rt = 0; rt < 4; ++rt)
#pragma unroll
        for (int r = 0; r < 8; ++r) {
            int m = m_base + rt * 16 + ((lane < 16) ? r : r + 8);
            if (m >= M) continue;
            int n = n0 + lm;
            float v = acc[rt][r] + bias[n];
            if (act == 1) v = silu(v);
            size_t o = (size_t)m * HDIM + n;
            if (resid) v += resid[o];
            if (outF) outF[o] = v;
            if (outB) outB[o] = (__bf16)v;
        }
}

// ---------------------------------------------------------------------------
// Neighbor embedding: W = (ea @ ne_dproj + b) * cm[e]; agg0[dst] += ne_emb[z[src]] * W
// Block = 64 edges. WMMA tiles: 4 row x 8 col, then LDS + atomic scatter.
// ---------------------------------------------------------------------------
__global__ __launch_bounds__(256) void k_ne_scatter(
    const __bf16* __restrict__ ea, const __bf16* __restrict__ Wp,
    const float* __restrict__ bias, const float* __restrict__ cm,
    const int* __restrict__ ei, const int* __restrict__ z,
    const float* __restrict__ ne_emb, float* __restrict__ agg) {
    __shared__ float Ws[64][HDIM];
    int lane = threadIdx.x & 31, wave = threadIdx.x >> 5;
    int hf = lane >> 4, lm = lane & 15;
    int n0 = wave * 16;
    int e0 = blockIdx.x * 64;

    v8f acc[4];
#pragma unroll
    for (int rt = 0; rt < 4; ++rt)
#pragma unroll
        for (int r = 0; r < 8; ++r) acc[rt][r] = 0.f;

#pragma unroll
    for (int k0 = 0; k0 < NRBFP; k0 += 32) {
        v16bf bf = load_b_packed(Wp, k0 >> 5, wave, lane);
#pragma unroll
        for (int rt = 0; rt < 4; ++rt) {
            v16bf af = load_a_frag(ea + (size_t)(e0 + rt * 16 + lm) * NRBFP + k0 + hf * 8);
            acc[rt] = wmma_bf16(af, bf, acc[rt]);
        }
    }
#pragma unroll
    for (int rt = 0; rt < 4; ++rt)
#pragma unroll
        for (int r = 0; r < 8; ++r) {
            int ml = rt * 16 + ((lane < 16) ? r : r + 8);
            int n = n0 + lm;
            Ws[ml][n] = (acc[rt][r] + bias[n]) * cm[e0 + ml];
        }
    __syncthreads();

    for (int i = threadIdx.x; i < 64 * HDIM; i += 256) {
        int el = i >> 7, c = i & 127;
        int e = e0 + el;
        int s = ei[e], d_ = ei[N_EDGES + e];
        float val = ne_emb[z[s] * HDIM + c] * Ws[el][c];
        atomicAdd(&agg[(size_t)d_ * HDIM + c], val);
    }
}

// ---------------------------------------------------------------------------
// Fused per-layer edge kernel: dk/dv = silu(f @ W + b) via WMMA (kept in LDS),
// attn = silu(sum_h q[dst]*k[src]*dk) * C, msg = v[src]*dv*attn scattered to agg.
// Block = 64 edges; 256 threads; 8 WMMAs per wave per 32-wide k-step.
// ---------------------------------------------------------------------------
__global__ __launch_bounds__(256) void k_edge_attn(
    const __bf16* __restrict__ fbf, const __bf16* __restrict__ wdk,
    const __bf16* __restrict__ wdv, const float* __restrict__ dkb,
    const float* __restrict__ dvb, const float* __restrict__ q,
    const float* __restrict__ kk, const float* __restrict__ vv,
    const float* __restrict__ C, const int* __restrict__ ei,
    float* __restrict__ agg) {
    __shared__ float dks[64][HDIM];
    __shared__ float dvs[64][HDIM];
    int lane = threadIdx.x & 31, wave = threadIdx.x >> 5;
    int hf = lane >> 4, lm = lane & 15;
    int n0 = wave * 16;
    int e0 = blockIdx.x * 64;

    v8f acc[2][4];
#pragma unroll
    for (int m2 = 0; m2 < 2; ++m2)
#pragma unroll
        for (int rt = 0; rt < 4; ++rt)
#pragma unroll
            for (int r = 0; r < 8; ++r) acc[m2][rt][r] = 0.f;

#pragma unroll
    for (int k0 = 0; k0 < HDIM; k0 += 32) {
        v16bf bk = load_b_packed(wdk, k0 >> 5, wave, lane);
        v16bf bv = load_b_packed(wdv, k0 >> 5, wave, lane);
#pragma unroll
        for (int rt = 0; rt < 4; ++rt) {
            v16bf af = load_a_frag(fbf + (size_t)(e0 + rt * 16 + lm) * HDIM + k0 + hf * 8);
            acc[0][rt] = wmma_bf16(af, bk, acc[0][rt]);
            acc[1][rt] = wmma_bf16(af, bv, acc[1][rt]);
        }
    }
#pragma unroll
    for (int rt = 0; rt < 4; ++rt)
#pragma unroll
        for (int r = 0; r < 8; ++r) {
            int ml = rt * 16 + ((lane < 16) ? r : r + 8);
            int n = n0 + lm;
            dks[ml][n] = silu(acc[0][rt][r] + dkb[n]);
            dvs[ml][n] = silu(acc[1][rt][r] + dvb[n]);
        }
    __syncthreads();

    // (edge, head) pairs: 64*8 = 512 -> 2 per thread
    for (int p = threadIdx.x; p < 64 * NHEADS; p += 256) {
        int el = p >> 3, h = p & 7;
        int e = e0 + el;
        int s = ei[e], d_ = ei[N_EDGES + e];
        const float* qp = q  + (size_t)d_ * HDIM + h * HEADD;
        const float* kp = kk + (size_t)s  * HDIM + h * HEADD;
        const float* vp = vv + (size_t)s  * HDIM + h * HEADD;
        float dot = 0.f;
#pragma unroll
        for (int i = 0; i < HEADD; ++i) dot += qp[i] * kp[i] * dks[el][h * HEADD + i];
        float attn = silu(dot) * C[e];
        float* ap = agg + (size_t)d_ * HDIM + h * HEADD;
#pragma unroll
        for (int i = 0; i < HEADD; ++i)
            atomicAdd(&ap[i], vp[i] * dvs[el][h * HEADD + i] * attn);
    }
}

// ---------------------------------------------------------------------------
// LayerNorm over H=128; one row per block (128 threads, 4 waves, shuffle+LDS reduce)
// ---------------------------------------------------------------------------
__global__ __launch_bounds__(128) void k_layernorm(
    const float* __restrict__ x, const float* __restrict__ g, const float* __restrict__ b,
    float* __restrict__ outF, __bf16* __restrict__ outB) {
    int row = blockIdx.x;
    int t = threadIdx.x;
    float v = x[(size_t)row * HDIM + t];
    float sum = v, sq = v * v;
#pragma unroll
    for (int off = 16; off > 0; off >>= 1) {
        sum += __shfl_down(sum, off, 32);
        sq  += __shfl_down(sq, off, 32);
    }
    __shared__ float s1[4], s2[4];
    if ((t & 31) == 0) { s1[t >> 5] = sum; s2[t >> 5] = sq; }
    __syncthreads();
    if (t == 0) {
        float a = 0.f, c = 0.f;
        for (int i = 0; i < 4; ++i) { a += s1[i]; c += s2[i]; }
        s1[0] = a; s2[0] = c;
    }
    __syncthreads();
    float mean = s1[0] * (1.f / HDIM);
    float var  = s2[0] * (1.f / HDIM) - mean * mean;
    float y = (v - mean) * rsqrtf(var + 1e-5f) * g[t] + b[t];
    if (outF) outF[(size_t)row * HDIM + t] = y;
    if (outB) outB[(size_t)row * HDIM + t] = (__bf16)y;
}

// ---------------------------------------------------------------------------
extern "C" void kernel_launch(void* const* d_in, const int* in_sizes, int n_in,
                              void* d_out, int out_size, void* d_ws, size_t ws_size,
                              hipStream_t stream) {
    const int*   z          = (const int*)d_in[0];
    const float* pos        = (const float*)d_in[1];
    const int*   ei         = (const int*)d_in[2];
    const float* emb        = (const float*)d_in[3];
    const float* ne_emb     = (const float*)d_in[4];
    const float* ne_dproj_w = (const float*)d_in[5];
    const float* ne_dproj_b = (const float*)d_in[6];
    const float* ne_comb_w  = (const float*)d_in[7];
    const float* ne_comb_b  = (const float*)d_in[8];
    const float* rbf_means  = (const float*)d_in[9];
    const float* rbf_betas  = (const float*)d_in[10];
    const float* rbf_proj_w = (const float*)d_in[11];
    const float* rbf_proj_b = (const float*)d_in[12];
    const float* ln_g       = (const float*)d_in[13];
    const float* ln_b       = (const float*)d_in[14];
    const float* qw = (const float*)d_in[15];
    const float* qb = (const float*)d_in[16];
    const float* kw = (const float*)d_in[17];
    const float* kb = (const float*)d_in[18];
    const float* vw = (const float*)d_in[19];
    const float* vb = (const float*)d_in[20];
    const float* ow = (const float*)d_in[21];
    const float* ob = (const float*)d_in[22];
    const float* dkw = (const float*)d_in[23];
    const float* dkb = (const float*)d_in[24];
    const float* dvw = (const float*)d_in[25];
    const float* dvb = (const float*)d_in[26];
    const float* out_ln_g = (const float*)d_in[27];
    const float* out_ln_b = (const float*)d_in[28];

    char* wsp = (char*)d_ws;
    auto carve = [&](size_t bytes) -> void* {
        void* p = (void*)wsp;
        wsp += (bytes + 255) & ~(size_t)255;
        return p;
    };
    __bf16* ea    = (__bf16*)carve((size_t)N_EDGES * NRBFP * 2);
    __bf16* fbf   = (__bf16*)carve((size_t)N_EDGES * HDIM * 2);
    float*  Cbuf  = (float*)carve((size_t)N_EDGES * 4);
    float*  cm    = (float*)carve((size_t)N_EDGES * 4);
    float*  x     = (float*)carve((size_t)N_NODES * HDIM * 4);
    __bf16* hbf   = (__bf16*)carve((size_t)N_NODES * HDIM * 2);
    float*  qn    = (float*)carve((size_t)N_NODES * HDIM * 4);
    float*  kn    = (float*)carve((size_t)N_NODES * HDIM * 4);
    float*  vn    = (float*)carve((size_t)N_NODES * HDIM * 4);
    float*  aggF  = (float*)carve((size_t)N_NODES * HDIM * 4);
    __bf16* aggB  = (__bf16*)carve((size_t)N_NODES * HDIM * 2);
    __bf16* xc    = (__bf16*)carve((size_t)N_NODES * 256 * 2);
    __bf16* w_dproj = (__bf16*)carve((size_t)NRBFP * HDIM * 2);
    __bf16* w_rproj = (__bf16*)carve((size_t)NRBFP * HDIM * 2);
    __bf16* w_comb  = (__bf16*)carve((size_t)256 * HDIM * 2);
    __bf16* wq_b  = (__bf16*)carve((size_t)NLAYERS * HDIM * HDIM * 2);
    __bf16* wk_b  = (__bf16*)carve((size_t)NLAYERS * HDIM * HDIM * 2);
    __bf16* wv_b  = (__bf16*)carve((size_t)NLAYERS * HDIM * HDIM * 2);
    __bf16* wo_b  = (__bf16*)carve((size_t)NLAYERS * HDIM * HDIM * 2);
    __bf16* wdk_b = (__bf16*)carve((size_t)NLAYERS * HDIM * HDIM * 2);
    __bf16* wdv_b = (__bf16*)carve((size_t)NLAYERS * HDIM * HDIM * 2);

    const int nodeElems = N_NODES * HDIM;
    const int gNode256  = (nodeElems + 255) / 256;
    const int gPad64    = (NRBFP * HDIM + 255) / 256;
    const int gPad128   = (HDIM * HDIM + 255) / 256;
    const int WELEMS    = HDIM * HDIM;

    // --- weight packing (fp32 -> WMMA-fragment-major bf16, zero-padded K) ---
    k_pack_b<<<gPad64, 256, 0, stream>>>(ne_dproj_w, w_dproj, NRBF, NRBFP);
    k_pack_b<<<gPad64, 256, 0, stream>>>(rbf_proj_w, w_rproj, NRBF, NRBFP);
    k_pack_b<<<(256 * HDIM + 255) / 256, 256, 0, stream>>>(ne_comb_w, w_comb, 256, 256);
    for (int l = 0; l < NLAYERS; ++l) {
        const size_t wl = (size_t)l * WELEMS;
        k_pack_b<<<gPad128, 256, 0, stream>>>(qw + wl,  wq_b + wl,  HDIM, HDIM);
        k_pack_b<<<gPad128, 256, 0, stream>>>(kw + wl,  wk_b + wl,  HDIM, HDIM);
        k_pack_b<<<gPad128, 256, 0, stream>>>(vw + wl,  wv_b + wl,  HDIM, HDIM);
        k_pack_b<<<gPad128, 256, 0, stream>>>(ow + wl,  wo_b + wl,  HDIM, HDIM);
        k_pack_b<<<gPad128, 256, 0, stream>>>(dkw + wl, wdk_b + wl, HDIM, HDIM);
        k_pack_b<<<gPad128, 256, 0, stream>>>(dvw + wl, wdv_b + wl, HDIM, HDIM);
    }

    // --- edge geometry + RBF ---
    k_edge_geom<<<(N_EDGES + 255) / 256, 256, 0, stream>>>(ei, pos, rbf_means, rbf_betas, ea, Cbuf, cm);

    // --- neighbor embedding: agg0 via WMMA + scatter ---
    k_zero<<<gNode256, 256, 0, stream>>>(aggF, nodeElems);
    k_embed<<<gNode256, 256, 0, stream>>>(z, emb, xc);
    k_ne_scatter<<<N_EDGES / 64, 256, 0, stream>>>(ea, w_dproj, ne_dproj_b, cm, ei, z, ne_emb, aggF);
    k_f32_to_bf16_strided<<<gNode256, 256, 0, stream>>>(aggF, xc, N_NODES, HDIM, 256, HDIM);

    // --- x = concat([emb[z], agg0]) @ ne_comb_w + b ---
    k_gemm128<<<(N_NODES + 63) / 64, 256, 0, stream>>>(xc, N_NODES, 256, w_comb, ne_comb_b,
                                                       nullptr, x, nullptr, 0);
    // --- f = ea @ rbf_proj_w + b (kept bf16 for per-layer WMMA) ---
    k_gemm128<<<N_EDGES / 64, 256, 0, stream>>>(ea, N_EDGES, NRBFP, w_rproj, rbf_proj_b,
                                                nullptr, nullptr, fbf, 0);

    // --- transformer layers ---
    for (int l = 0; l < NLAYERS; ++l) {
        const size_t wl = (size_t)l * WELEMS;
        k_layernorm<<<N_NODES, 128, 0, stream>>>(x, ln_g + l * HDIM, ln_b + l * HDIM, nullptr, hbf);
        k_gemm128<<<(N_NODES + 63) / 64, 256, 0, stream>>>(hbf, N_NODES, HDIM, wq_b + wl,
                                                           qb + l * HDIM, nullptr, qn, nullptr, 0);
        k_gemm128<<<(N_NODES + 63) / 64, 256, 0, stream>>>(hbf, N_NODES, HDIM, wk_b + wl,
                                                           kb + l * HDIM, nullptr, kn, nullptr, 0);
        k_gemm128<<<(N_NODES + 63) / 64, 256, 0, stream>>>(hbf, N_NODES, HDIM, wv_b + wl,
                                                           vb + l * HDIM, nullptr, vn, nullptr, 0);
        k_zero<<<gNode256, 256, 0, stream>>>(aggF, nodeElems);
        k_edge_attn<<<N_EDGES / 64, 256, 0, stream>>>(fbf, wdk_b + wl, wdv_b + wl,
                                                      dkb + l * HDIM, dvb + l * HDIM,
                                                      qn, kn, vn, Cbuf, ei, aggF);
        k_f32_to_bf16_strided<<<gNode256, 256, 0, stream>>>(aggF, aggB, N_NODES, HDIM, HDIM, 0);
        k_gemm128<<<(N_NODES + 63) / 64, 256, 0, stream>>>(aggB, N_NODES, HDIM, wo_b + wl,
                                                           ob + l * HDIM, x, x, nullptr, 0);
    }

    // --- output layernorm ---
    k_layernorm<<<N_NODES, 128, 0, stream>>>(x, out_ln_g, out_ln_b, (float*)d_out, nullptr);
}
